// StandardRnn_76510547410999
// MI455X (gfx1250) — matrix operand order repaired and checked by reading, not compiled, not run on hardware
//
#include <hip/hip_runtime.h>
#include <hip/hip_bf16.h>

typedef __attribute__((ext_vector_type(16))) __bf16 bf16x16;
typedef __attribute__((ext_vector_type(8)))  __bf16 bf16x8;
typedef __attribute__((ext_vector_type(8)))  float  f32x8;

#define B_   32
#define S_   1024
#define D_   512
#define H_   1024
#define NWG  16
#define CW   64   // hidden columns owned per workgroup (16 * 64 = 1024)

// CDNA5 async global->LDS copy (ASYNCcnt path). LDS dest address is the low
// 32 bits of the generic pointer (flat aperture: addr[31:0] = LDS offset).
__device__ __forceinline__ void async_b128_to_lds(unsigned lds_off, const void* gaddr) {
    asm volatile("global_load_async_to_lds_b128 %0, %1, off"
                 :: "v"(lds_off), "v"(gaddr) : "memory");
}
__device__ __forceinline__ void wait_async0() {
    asm volatile("s_wait_asynccnt 0" ::: "memory");
}

// ---------------------------------------------------------------------------
// Kernel 0: prep. Transpose+convert weights to bf16, zero h0 and the barrier
// counter (every launch -> graph-replay deterministic).
// ---------------------------------------------------------------------------
__global__ __launch_bounds__(256) void rnn_prep_kernel(
    const float* __restrict__ Wih, const float* __restrict__ Whh,
    __bf16* __restrict__ WihT, __bf16* __restrict__ WhhT,
    __bf16* __restrict__ h0, unsigned int* __restrict__ counter)
{
    unsigned int i = blockIdx.x * 256u + threadIdx.x;
    if (i < 1024u * 512u) {            // WihT[n][k] = Wih[k][n]
        int n = i >> 9, k = i & 511;
        WihT[i] = (__bf16)Wih[(size_t)k * H_ + n];
    }
    if (i < 1024u * 1024u) {           // WhhT[n][k] = Whh[k][n]
        int n = i >> 10, k = i & 1023;
        WhhT[i] = (__bf16)Whh[(size_t)k * H_ + n];
    }
    if (i < (unsigned)(B_ * H_)) h0[i] = (__bf16)0.0f;
    if (i == 0) *counter = 0u;
}

// ---------------------------------------------------------------------------
// Kernel 1: xp = x @ W_ih + b -> d_out (scan runs in place over it).
// One 16x16 tile per wave; A fragment packed from fp32 x via v_cvt_pk_bf16,
// B fragment = 32B contiguous-K load from pre-transposed bf16 weights.
// ---------------------------------------------------------------------------
__global__ __launch_bounds__(256) void xproj_kernel(
    const float* __restrict__ x, const __bf16* __restrict__ WihT,
    const float* __restrict__ bias, float* __restrict__ out)
{
    const int lane = threadIdx.x & 31;
    const int gw   = blockIdx.x * 8 + (threadIdx.x >> 5);
    const int mi   = gw >> 6;      // 0..2047 row tiles
    const int ni   = gw & 63;      // 0..63  col tiles
    const int m15  = lane & 15;
    const int hiK  = (lane >> 4) & 1;
    const int arow = mi * 16 + m15;
    const int ncol = ni * 16 + m15;
    const float*  xr = x    + (size_t)arow * D_;
    const __bf16* wr = WihT + (size_t)ncol * D_;
    const int ka = hiK * 16;
    const int kb = hiK * 16;

    f32x8 acc = {};
    for (int k0 = 0; k0 < D_; k0 += 32) {
        float4 a0 = *(const float4*)(xr + k0 + ka);
        float4 a1 = *(const float4*)(xr + k0 + ka + 4);
        float4 a2 = *(const float4*)(xr + k0 + ka + 8);
        float4 a3 = *(const float4*)(xr + k0 + ka + 12);
        bf16x16 a;
        a[0]=(__bf16)a0.x;  a[1]=(__bf16)a0.y;  a[2]=(__bf16)a0.z;  a[3]=(__bf16)a0.w;
        a[4]=(__bf16)a1.x;  a[5]=(__bf16)a1.y;  a[6]=(__bf16)a1.z;  a[7]=(__bf16)a1.w;
        a[8]=(__bf16)a2.x;  a[9]=(__bf16)a2.y;  a[10]=(__bf16)a2.z; a[11]=(__bf16)a2.w;
        a[12]=(__bf16)a3.x; a[13]=(__bf16)a3.y; a[14]=(__bf16)a3.z; a[15]=(__bf16)a3.w;
        bf16x16 bm = *(const bf16x16*)(wr + k0 + kb);
        acc = __builtin_amdgcn_wmma_f32_16x16x32_bf16(
            false, a, false, bm, (short)0, acc, false, false);
    }
    const float bv    = bias[ncol];
    const int   mbase = mi * 16 + hiK * 8;
    #pragma unroll
    for (int r = 0; r < 8; ++r)
        out[(size_t)(mbase + r) * H_ + ncol] = acc[r] + bv;
}

// ---------------------------------------------------------------------------
// Kernel 2: persistent scan. 16 WGs x 256 threads. 192 KB LDS per WG:
//   chunks [0..4095]     : h_prev (32x1024 bf16), 16B-chunk XOR swizzle
//   chunks [4096..12287] : W_hh slice (64 cols x 1024 K bf16), 32B XOR swizzle
// W_hh slice staged once, h restaged per step — both via async global->LDS.
// 32 bf16 WMMAs/wave/step, tanh epilogue, atomic grid barrier per step.
// ---------------------------------------------------------------------------
__global__ __launch_bounds__(256, 1) void rnn_scan_kernel(
    const __bf16* __restrict__ WhhT,
    __bf16* __restrict__ hb0, __bf16* __restrict__ hb1,
    float* __restrict__ out, unsigned int* __restrict__ counter)
{
    __shared__ uint4 lds[12288];   // 192 KB

    const int tid  = threadIdx.x;
    const int lane = tid & 31;
    const int wave = tid >> 5;
    const int mi   = wave >> 2;            // 0..1  row tile
    const int ni   = wave & 3;             // 0..3  col tile in slice
    const int m15  = lane & 15;
    const int hiK  = (lane >> 4) & 1;
    const int c0   = blockIdx.x * CW;
    const int ncol = c0 + ni * 16 + m15;   // global hidden column
    const int arow = mi * 16 + m15;        // batch row (A fragment)
    const int sw   = arow & 15;            // A chunk swizzle key
    const int lnrow = ni * 16 + m15;       // local W_hh row (B fragment)
    const int swb   = lnrow & 15;          // B granule swizzle key
    const int mrowbase = mi * 16 + hiK * 8;

    // --- one-time: async-stage W_hh slice (cols [c0,c0+CW), all K) ---
    {
        const __bf16* src = WhhT + (size_t)c0 * H_;
        #pragma unroll
        for (int it = 0; it < 32; ++it) {
            int qc = it * 256 + tid;       // 16B chunk id 0..8191
            int ln = qc >> 7;              // slice row 0..63
            int cc = qc & 127;             // chunk within row
            int g  = cc >> 1, par = cc & 1;
            unsigned dst = (unsigned)(uintptr_t)
                &lds[4096 + (ln << 7) + (((g ^ (ln & 15)) << 1) | par)];
            async_b128_to_lds(dst, src + ((size_t)ln << 10) + (cc << 3));
        }
    }

    __bf16* hbuf[2] = { hb0, hb1 };

    for (int t = 0; t < S_; ++t) {
        // --- async-stage h_prev -> LDS (swizzled 16B chunks) ---
        const __bf16* hsrc = hbuf[t & 1];
        #pragma unroll
        for (int it = 0; it < 16; ++it) {
            int q = it * 256 + tid;        // chunk id 0..4095
            int m = q >> 7, cch = q & 127;
            unsigned dst = (unsigned)(uintptr_t)&lds[(m << 7) | (cch ^ (m & 15))];
            async_b128_to_lds(dst, hsrc + (q << 3));
        }
        wait_async0();
        __syncthreads();

        // --- h @ W_hh[:, cols] via WMMA, K = 1024, all operands in LDS ---
        f32x8 acc = {};
        for (int k0 = 0; k0 < H_; k0 += 32) {
            int ca = (k0 >> 3) + 2 * hiK;
            bf16x8 a0 = *(const bf16x8*)&lds[(arow << 7) | ( ca      ^ sw)];
            bf16x8 a1 = *(const bf16x8*)&lds[(arow << 7) | ((ca + 1) ^ sw)];
            bf16x16 a;
            #pragma unroll
            for (int e = 0; e < 8; ++e) { a[e] = a0[e]; a[e + 8] = a1[e]; }
            bf16x16 bm = *(const bf16x16*)
                &lds[4096 + (lnrow << 7) + ((((k0 >> 4) + hiK) ^ swb) << 1)];
            acc = __builtin_amdgcn_wmma_f32_16x16x32_bf16(
                false, a, false, bm, (short)0, acc, false, false);
        }

        // --- epilogue: tanh(acc + xp), write out + next-h (bf16) ---
        __bf16* hdst = hbuf[(t + 1) & 1];
        #pragma unroll
        for (int r = 0; r < 8; ++r) {
            int m = mrowbase + r;
            size_t oidx = ((size_t)m * S_ + t) * H_ + ncol;
            float v = tanhf(acc[r] + out[oidx]);
            out[oidx] = v;
            hdst[m * H_ + ncol] = (__bf16)v;
        }

        // --- grid-wide step barrier (monotone counter) ---
        __syncthreads();
        if (tid == 0) {
            __threadfence();
            atomicAdd(counter, 1u);
            unsigned goal = (unsigned)(NWG * (t + 1));
            while (__hip_atomic_load(counter, __ATOMIC_RELAXED,
                                     __HIP_MEMORY_SCOPE_AGENT) < goal) {
                __builtin_amdgcn_s_sleep(1);
            }
            __threadfence();
        }
        __syncthreads();
    }
}

// ---------------------------------------------------------------------------
extern "C" void kernel_launch(void* const* d_in, const int* in_sizes, int n_in,
                              void* d_out, int out_size, void* d_ws, size_t ws_size,
                              hipStream_t stream) {
    const float* x    = (const float*)d_in[0];   // [B,S,D]
    const float* Wih  = (const float*)d_in[1];   // [D,H]
    const float* Whh  = (const float*)d_in[2];   // [H,H]
    const float* bias = (const float*)d_in[3];   // [H]
    float* out = (float*)d_out;                  // [B,S,H]

    char* ws = (char*)d_ws;
    __bf16* WihT = (__bf16*)ws;                              // 1 MB  [H][D] bf16
    __bf16* WhhT = (__bf16*)(ws + (1u << 20));               // 2 MB  [H][H] bf16
    __bf16* hb0  = (__bf16*)(ws + 3u * (1u << 20));          // 64 KB [B][H] bf16
    __bf16* hb1  = (__bf16*)(ws + 3u * (1u << 20) + 65536);  // 64 KB
    unsigned int* counter = (unsigned int*)(ws + 3u * (1u << 20) + 2 * 65536);

    rnn_prep_kernel<<<4096, 256, 0, stream>>>(Wih, Whh, WihT, WhhT, hb0, counter);
    xproj_kernel<<<16384, 256, 0, stream>>>(x, WihT, bias, out);
    rnn_scan_kernel<<<NWG, 256, 0, stream>>>(WhhT, hb0, hb1, out, counter);
}
